// DFF_67619965108631
// MI455X (gfx1250) — matrix operand order repaired and checked by reading, not compiled, not run on hardware
//
#include <hip/hip_runtime.h>
#include <hip/hip_bf16.h>
#include <math.h>

#define H_DIM  256
#define STAR_D 64
#define OUT_D  128

typedef __attribute__((ext_vector_type(16))) __bf16 v16bf;
typedef __attribute__((ext_vector_type(8)))  float  v8f;

__device__ __forceinline__ float lrelu_f(float x) { return x > 0.f ? x : 0.2f * x; }
__device__ __forceinline__ float elu_f(float x)   { return x > 0.f ? x : (expf(x) - 1.f); }

// order-preserving float -> uint key for atomicMax over signed floats
__device__ __forceinline__ unsigned ordkey(float x) {
    unsigned u = __float_as_uint(x);
    return (u & 0x80000000u) ? ~u : (u | 0x80000000u);
}
__device__ __forceinline__ float ordkey_inv(unsigned k) {
    unsigned u = (k & 0x80000000u) ? (k & 0x7fffffffu) : ~k;
    return __uint_as_float(u);
}

// ---------------------------------------------------------------------------
// Weight pre-pack: split fp32 W[K x Ncols] into bf16 hi/lo, laid out
// fragment-native for V_WMMA_F32_16X16X32_BF16's B operand:
//   dst[ ((k/32)*Ncols + col)*32 + (k%32) ]
// so a lane's 16 B-fragment elements (K = kb..kb+15 for one column) are
// 32 contiguous bytes -> one v16bf vector load in the GEMM.
// ---------------------------------------------------------------------------
__global__ void k_prepack(const float* __restrict__ W,
                          __bf16* __restrict__ hi, __bf16* __restrict__ lo,
                          int K, int Ncols)
{
    int idx = blockIdx.x * blockDim.x + threadIdx.x;
    if (idx >= K * Ncols) return;
    int k = idx / Ncols;
    int c = idx - k * Ncols;
    float  x = W[idx];
    __bf16 h = (__bf16)x;
    float  l = x - (float)h;
    size_t d = ((size_t)(k >> 5) * Ncols + c) * 32 + (k & 31);
    hi[d] = h;
    lo[d] = (__bf16)l;
}

// ---------------------------------------------------------------------------
// GEMM: C[rows x Ncols] = A[rows x K] @ W[K x Ncols] + bias
// bf16x3 split-precision on V_WMMA_F32_16X16X32_BF16 (~fp32 accuracy).
// Block = 128 threads = 4 waves; each wave computes a 16x64 strip.
// rows % 16 == 0; K % 32 == 0; Ncols % 64 == 0.
// ---------------------------------------------------------------------------
__global__ __launch_bounds__(128) void k_gemm(
    const float* __restrict__ A, int lda,
    const __bf16* __restrict__ Wh, const __bf16* __restrict__ Wl,
    const float* __restrict__ bias,
    float* __restrict__ C, int ldc,
    int rows, int K, int Ncols)
{
    const int lane = threadIdx.x & 31;
    const int wid  = threadIdx.x >> 5;
    const int r0   = blockIdx.x * 64 + wid * 16;
    if (r0 >= rows) return;              // wave-uniform: EXEC stays all-ones
    const int c0   = blockIdx.y * 64;

    const int m    = lane & 15;
    const int kbA  = (lane < 16) ? 0 : 16;   // A 16x32 bf16: contiguous k = 16h+s (matches B)
    const int kbB  = (lane < 16) ? 0 : 16;   // B 32x16 bf16 layout
    const int colB = c0 + m;

    const float* arow = A + (size_t)(r0 + m) * lda + kbA;

    v8f acc[4] = {};

    for (int k0 = 0; k0 < K; k0 += 32) {
        const int kt = k0 >> 5;
        // --- issue all loads for this k-step up front ---
        float4 a0 = *(const float4*)(arow + k0);
        float4 a1 = *(const float4*)(arow + k0 + 4);
        float4 a2 = *(const float4*)(arow + k0 + 8);
        float4 a3 = *(const float4*)(arow + k0 + 12);

        const size_t off = ((size_t)kt * Ncols + colB) * 32 + kbB;
        v16bf bh[4], bl[4];
#pragma unroll
        for (int t = 0; t < 4; ++t) {
            bh[t] = *(const v16bf*)(Wh + off + (size_t)t * 16 * 32);
            bl[t] = *(const v16bf*)(Wl + off + (size_t)t * 16 * 32);
        }

        // --- split A into bf16 hi/lo ---
        float av[16];
        ((float4*)av)[0] = a0;
        ((float4*)av)[1] = a1;
        ((float4*)av)[2] = a2;
        ((float4*)av)[3] = a3;
        v16bf ah, al;
#pragma unroll
        for (int e = 0; e < 16; ++e) {
            float  x  = av[e];
            __bf16 h_ = (__bf16)x;
            float  l_ = x - (float)h_;
            ah[e] = h_;
            al[e] = (__bf16)l_;
        }

        // --- 12 WMMAs: accH*bH + aH*bL + aL*bH into fp32 accumulators ---
#pragma unroll
        for (int t = 0; t < 4; ++t) {
            acc[t] = __builtin_amdgcn_wmma_f32_16x16x32_bf16(false, ah, false, bh[t], (short)0, acc[t], false, false);
            acc[t] = __builtin_amdgcn_wmma_f32_16x16x32_bf16(false, ah, false, bl[t], (short)0, acc[t], false, false);
            acc[t] = __builtin_amdgcn_wmma_f32_16x16x32_bf16(false, al, false, bh[t], (short)0, acc[t], false, false);
        }
    }

    const int rbase = r0 + ((lane < 16) ? 0 : 8);
#pragma unroll
    for (int t = 0; t < 4; ++t) {
        int   col = c0 + 16 * t + m;
        float b   = bias[col];
#pragma unroll
        for (int v = 0; v < 8; ++v)
            C[(size_t)(rbase + v) * ldc + col] = acc[t][v] + b;
    }
}

// ---------------------------------------------------------------------------
// Segment / elementwise kernels
// ---------------------------------------------------------------------------
__global__ void k_deg(const int* __restrict__ V, const int* __restrict__ E,
                      float* degv, float* dege, int nnz)
{
    int i = blockIdx.x * blockDim.x + threadIdx.x;
    if (i >= nnz) return;
    atomicAdd(&degv[V[i]], 1.f);
    atomicAdd(&dege[E[i]], 1.f);
}

// one wave32 per node: score = X_feat[n] . a
__global__ void k_score(const float* __restrict__ Xf, const float* __restrict__ a,
                        float* __restrict__ out, int N, int H)
{
    int wave = (int)((blockIdx.x * (size_t)blockDim.x + threadIdx.x) >> 5);
    int lane = threadIdx.x & 31;
    if (wave >= N) return;
    const float* row = Xf + (size_t)wave * H;
    float s = 0.f;
    for (int c = lane; c < H; c += 32) s += row[c] * a[c];
    for (int off = 16; off > 0; off >>= 1) s += __shfl_down(s, off, 32);
    if (lane == 0) out[wave] = s;
}

__global__ void k_edge_max(const float* __restrict__ score, const int* __restrict__ V,
                           const int* __restrict__ E, unsigned* __restrict__ emax, int nnz)
{
    int i = blockIdx.x * blockDim.x + threadIdx.x;
    if (i >= nnz) return;
    float s = lrelu_f(score[V[i]]);
    atomicMax(&emax[E[i]], ordkey(s));
}

__global__ void k_exp(const float* __restrict__ score, const int* __restrict__ V,
                      const int* __restrict__ E, const unsigned* __restrict__ emax,
                      float* __restrict__ ex, float* __restrict__ denom, int nnz)
{
    int i = blockIdx.x * blockDim.x + threadIdx.x;
    if (i >= nnz) return;
    int   e  = E[i];
    float mx = ordkey_inv(emax[e]);
    float s  = lrelu_f(score[V[i]]);
    float ev = expf(s - mx);
    ex[i] = ev;
    atomicAdd(&denom[e], ev);
}

// one 64-thread block per nnz entry; float4 gather + 4 scalar atomic adds
__global__ __launch_bounds__(64) void k_scatter_v2e(
    const float* __restrict__ Xf, const int* __restrict__ V, const int* __restrict__ E,
    const float* __restrict__ ex, const float* __restrict__ denom,
    float* __restrict__ Yv2e, int nnz)
{
    int b = blockIdx.x;
    if (b >= nnz) return;
    int t = threadIdx.x;
    int v = V[b], e = E[b];
    float w = ex[b] / denom[e];
    float4 x = ((const float4*)(Xf + (size_t)v * H_DIM))[t];
    float* dst = Yv2e + (size_t)e * H_DIM + t * 4;
    atomicAdd(dst + 0, x.x * w);
    atomicAdd(dst + 1, x.y * w);
    atomicAdd(dst + 2, x.z * w);
    atomicAdd(dst + 3, x.w * w);
}

// e_msg = [ elu(Yv2e) | S ]  shaped [M, H+STAR]
__global__ void k_emsg(const float* __restrict__ Yv2e, const float* __restrict__ S,
                       float* __restrict__ emsg, int M)
{
    int idx = blockIdx.x * blockDim.x + threadIdx.x;
    int total = M * (H_DIM + STAR_D);
    if (idx >= total) return;
    int e = idx / (H_DIM + STAR_D);
    int c = idx - e * (H_DIM + STAR_D);
    emsg[idx] = (c < H_DIM) ? elu_f(Yv2e[(size_t)e * H_DIM + c])
                            : S[(size_t)e * STAR_D + (c - H_DIM)];
}

__global__ __launch_bounds__(64) void k_scatter_e2n(
    const float* __restrict__ Y, const int* __restrict__ V, const int* __restrict__ E,
    float* __restrict__ Xn_sum, int nnz)
{
    int b = blockIdx.x;
    if (b >= nnz) return;
    int t = threadIdx.x;
    int v = V[b], e = E[b];
    float4 y = ((const float4*)(Y + (size_t)e * H_DIM))[t];
    float* dst = Xn_sum + (size_t)v * H_DIM + t * 4;
    atomicAdd(dst + 0, y.x);
    atomicAdd(dst + 1, y.y);
    atomicAdd(dst + 2, y.z);
    atomicAdd(dst + 3, y.w);
}

// h = elu(seg_mean) + X_init  (in place into X_init buffer)
__global__ void k_finalize(const float* __restrict__ Xn_sum, const float* __restrict__ degv,
                           float* __restrict__ Xinit, int total)
{
    int idx = blockIdx.x * blockDim.x + threadIdx.x;
    if (idx >= total) return;
    int   n    = idx >> 8;   // H_DIM == 256
    float cnt  = degv[n];
    float mean = (cnt > 0.f) ? Xn_sum[idx] / fmaxf(cnt, 1.f) : 0.f;
    Xinit[idx] = elu_f(mean) + Xinit[idx];
}

// hyperconv: scatter node features + node degrees to edges
__global__ __launch_bounds__(32) void k_hc_scatter1(
    const float* __restrict__ Xn2, const int* __restrict__ V, const int* __restrict__ E,
    const float* __restrict__ degv, float* __restrict__ Ysum, float* __restrict__ sumDv, int nnz)
{
    int b = blockIdx.x;
    if (b >= nnz) return;
    int t = threadIdx.x;
    int v = V[b], e = E[b];
    float4 x = ((const float4*)(Xn2 + (size_t)v * OUT_D))[t];
    float* dst = Ysum + (size_t)e * OUT_D + t * 4;
    atomicAdd(dst + 0, x.x);
    atomicAdd(dst + 1, x.y);
    atomicAdd(dst + 2, x.z);
    atomicAdd(dst + 3, x.w);
    if (t == 0) atomicAdd(&sumDv[e], degv[v]);
}

// Ye = De^-1/2 * seg_mean  (in place)
__global__ void k_hc_edge(float* __restrict__ Ysum, const float* __restrict__ dege,
                          const float* __restrict__ sumDv, int total)
{
    int idx = blockIdx.x * blockDim.x + threadIdx.x;
    if (idx >= total) return;
    int   e    = idx >> 7;   // OUT_D == 128
    float cnt  = dege[e];
    float mean = (cnt > 0.f) ? Ysum[idx] / fmaxf(cnt, 1.f) : 0.f;
    float De   = sumDv[e] / (cnt + 1.f);
    float dinv = (De > 0.f) ? rsqrtf(De) : 1.f;
    Ysum[idx] = dinv * mean;
}

__global__ __launch_bounds__(32) void k_hc_scatter2(
    const float* __restrict__ Ye, const int* __restrict__ V, const int* __restrict__ E,
    float* __restrict__ out_sum, int nnz)
{
    int b = blockIdx.x;
    if (b >= nnz) return;
    int t = threadIdx.x;
    int v = V[b], e = E[b];
    float4 y = ((const float4*)(Ye + (size_t)e * OUT_D))[t];
    float* dst = out_sum + (size_t)v * OUT_D + t * 4;
    atomicAdd(dst + 0, y.x);
    atomicAdd(dst + 1, y.y);
    atomicAdd(dst + 2, y.z);
    atomicAdd(dst + 3, y.w);
}

__global__ void k_out(const float* __restrict__ out_sum, const float* __restrict__ degv,
                      float* __restrict__ out, int total)
{
    int idx = blockIdx.x * blockDim.x + threadIdx.x;
    if (idx >= total) return;
    int   n = idx >> 7;
    float d = degv[n];
    out[idx] = (d > 0.f) ? rsqrtf(d) * out_sum[idx] : 0.f;
}

// ---------------------------------------------------------------------------
// Host orchestration
// ---------------------------------------------------------------------------
extern "C" void kernel_launch(void* const* d_in, const int* in_sizes, int n_in,
                              void* d_out, int out_size, void* d_ws, size_t ws_size,
                              hipStream_t stream)
{
    (void)n_in; (void)out_size; (void)ws_size;

    const float* X   = (const float*)d_in[0];
    const int*   V   = (const int*)d_in[1];
    const int*   E   = (const int*)d_in[2];
    const float* S   = (const float*)d_in[3];
    const float* Wx0 = (const float*)d_in[4];
    const float* bx0 = (const float*)d_in[5];
    const float* Wv0 = (const float*)d_in[6];
    const float* bv0 = (const float*)d_in[7];
    const float* a0  = (const float*)d_in[8];
    const float* Wt0 = (const float*)d_in[9];
    const float* bt0 = (const float*)d_in[10];
    const float* Wx1 = (const float*)d_in[11];
    const float* bx1 = (const float*)d_in[12];
    const float* Wv1 = (const float*)d_in[13];
    const float* bv1 = (const float*)d_in[14];
    const float* a1  = (const float*)d_in[15];
    const float* Wt1 = (const float*)d_in[16];
    const float* bt1 = (const float*)d_in[17];
    const float* Wf  = (const float*)d_in[18];
    const float* bf  = (const float*)d_in[19];

    const int IN  = 128;
    const int N   = in_sizes[0] / IN;      // 50000
    const int NNZ = in_sizes[1];           // 400000
    const int M   = in_sizes[3] / STAR_D;  // 20000

    // workspace carve-out (256B aligned)
    char* p = (char*)d_ws;
    auto carve = [&](size_t bytes) -> void* {
        void* r = (void*)p;
        p += (bytes + 255) & ~(size_t)255;
        return r;
    };
    float*    bufA   = (float*)carve((size_t)N * H_DIM * 4);        // X_init L0 -> h0
    float*    bufB   = (float*)carve((size_t)N * H_DIM * 4);        // X_feat (both layers)
    float*    bufC   = (float*)carve((size_t)N * H_DIM * 4);        // Xn_sum (both layers)
    float*    bufD   = (float*)carve((size_t)N * H_DIM * 4);        // X_init L1 -> h1
    float*    score  = (float*)carve((size_t)N * 4);
    float*    ex     = (float*)carve((size_t)NNZ * 4);
    unsigned* emax   = (unsigned*)carve((size_t)M * 4);
    float*    denom  = (float*)carve((size_t)M * 4);
    float*    yv2e   = (float*)carve((size_t)M * H_DIM * 4);
    float*    emsg   = (float*)carve((size_t)M * (H_DIM + STAR_D) * 4);
    float*    ybuf   = (float*)carve((size_t)M * H_DIM * 4);
    float*    degv   = (float*)carve((size_t)N * 4);
    float*    dege   = (float*)carve((size_t)M * 4);
    float*    sumdv  = (float*)carve((size_t)M * 4);
    float*    xn2    = (float*)carve((size_t)N * OUT_D * 4);
    float*    ysum   = (float*)carve((size_t)M * OUT_D * 4);
    float*    outsum = (float*)carve((size_t)N * OUT_D * 4);

    // packed bf16 hi/lo weights
    const int KX0 = IN,   KT = H_DIM + STAR_D, KH = H_DIM;
    __bf16* pWx0h = (__bf16*)carve((size_t)KX0 * H_DIM * 2);
    __bf16* pWx0l = (__bf16*)carve((size_t)KX0 * H_DIM * 2);
    __bf16* pWv0h = (__bf16*)carve((size_t)KX0 * H_DIM * 2);
    __bf16* pWv0l = (__bf16*)carve((size_t)KX0 * H_DIM * 2);
    __bf16* pWt0h = (__bf16*)carve((size_t)KT * H_DIM * 2);
    __bf16* pWt0l = (__bf16*)carve((size_t)KT * H_DIM * 2);
    __bf16* pWx1h = (__bf16*)carve((size_t)KH * H_DIM * 2);
    __bf16* pWx1l = (__bf16*)carve((size_t)KH * H_DIM * 2);
    __bf16* pWv1h = (__bf16*)carve((size_t)KH * H_DIM * 2);
    __bf16* pWv1l = (__bf16*)carve((size_t)KH * H_DIM * 2);
    __bf16* pWt1h = (__bf16*)carve((size_t)KT * H_DIM * 2);
    __bf16* pWt1l = (__bf16*)carve((size_t)KT * H_DIM * 2);
    __bf16* pWfh  = (__bf16*)carve((size_t)KH * OUT_D * 2);
    __bf16* pWfl  = (__bf16*)carve((size_t)KH * OUT_D * 2);

    auto prep = [&](const float* W, __bf16* hi, __bf16* lo, int K, int Nc) {
        int tot = K * Nc;
        k_prepack<<<(tot + 255) / 256, 256, 0, stream>>>(W, hi, lo, K, Nc);
    };
    prep(Wx0, pWx0h, pWx0l, KX0, H_DIM);
    prep(Wv0, pWv0h, pWv0l, KX0, H_DIM);
    prep(Wt0, pWt0h, pWt0l, KT,  H_DIM);
    prep(Wx1, pWx1h, pWx1l, KH,  H_DIM);
    prep(Wv1, pWv1h, pWv1l, KH,  H_DIM);
    prep(Wt1, pWt1h, pWt1l, KT,  H_DIM);
    prep(Wf,  pWfh,  pWfl,  KH,  OUT_D);

    auto gemm = [&](const float* A, int lda, const __bf16* Wh, const __bf16* Wl,
                    const float* bias, float* C, int rows, int K, int Ncols) {
        dim3 g((rows + 63) / 64, Ncols / 64);
        k_gemm<<<g, dim3(128), 0, stream>>>(A, lda, Wh, Wl, bias, C, Ncols, rows, K, Ncols);
    };

    // degrees (once)
    hipMemsetAsync(degv, 0, (size_t)N * 4, stream);
    hipMemsetAsync(dege, 0, (size_t)M * 4, stream);
    k_deg<<<(NNZ + 255) / 256, 256, 0, stream>>>(V, E, degv, dege, NNZ);

    auto layer = [&](const float* Xin, int Kin,
                     const __bf16* pWxh, const __bf16* pWxl, const float* bx,
                     const __bf16* pWvh, const __bf16* pWvl, const float* bv,
                     const float* a,
                     const __bf16* pWth, const __bf16* pWtl, const float* bt,
                     float* initOut) {
        gemm(Xin, Kin, pWxh, pWxl, bx, initOut, N, Kin, H_DIM);    // X_init
        gemm(Xin, Kin, pWvh, pWvl, bv, bufB, N, Kin, H_DIM);       // X_feat
        k_score<<<(N + 3) / 4, 128, 0, stream>>>(bufB, a, score, N, H_DIM);
        hipMemsetAsync(emax, 0, (size_t)M * 4, stream);
        k_edge_max<<<(NNZ + 255) / 256, 256, 0, stream>>>(score, V, E, emax, NNZ);
        hipMemsetAsync(denom, 0, (size_t)M * 4, stream);
        k_exp<<<(NNZ + 255) / 256, 256, 0, stream>>>(score, V, E, emax, ex, denom, NNZ);
        hipMemsetAsync(yv2e, 0, (size_t)M * H_DIM * 4, stream);
        k_scatter_v2e<<<NNZ, 64, 0, stream>>>(bufB, V, E, ex, denom, yv2e, NNZ);
        int emsg_total = M * (H_DIM + STAR_D);
        k_emsg<<<(emsg_total + 255) / 256, 256, 0, stream>>>(yv2e, S, emsg, M);
        gemm(emsg, H_DIM + STAR_D, pWth, pWtl, bt, ybuf, M, H_DIM + STAR_D, H_DIM);
        hipMemsetAsync(bufC, 0, (size_t)N * H_DIM * 4, stream);
        k_scatter_e2n<<<NNZ, 64, 0, stream>>>(ybuf, V, E, bufC, NNZ);
        int ntot = N * H_DIM;
        k_finalize<<<(ntot + 255) / 256, 256, 0, stream>>>(bufC, degv, initOut, ntot);
    };

    layer(X,    IN,    pWx0h, pWx0l, bx0, pWv0h, pWv0l, bv0, a0, pWt0h, pWt0l, bt0, bufA);
    layer(bufA, H_DIM, pWx1h, pWx1l, bx1, pWv1h, pWv1l, bv1, a1, pWt1h, pWt1l, bt1, bufD);

    // hyperconv
    gemm(bufD, H_DIM, pWfh, pWfl, bf, xn2, N, H_DIM, OUT_D);
    hipMemsetAsync(ysum, 0, (size_t)M * OUT_D * 4, stream);
    hipMemsetAsync(sumdv, 0, (size_t)M * 4, stream);
    k_hc_scatter1<<<NNZ, 32, 0, stream>>>(xn2, V, E, degv, ysum, sumdv, NNZ);
    int etot = M * OUT_D;
    k_hc_edge<<<(etot + 255) / 256, 256, 0, stream>>>(ysum, dege, sumdv, etot);
    hipMemsetAsync(outsum, 0, (size_t)N * OUT_D * 4, stream);
    k_hc_scatter2<<<NNZ, 32, 0, stream>>>(ysum, V, E, outsum, NNZ);
    int otot = N * OUT_D;
    k_out<<<(otot + 255) / 256, 256, 0, stream>>>(outsum, degv, (float*)d_out, otot);
}
